// SurfaceGraphCommunication_50182397887126
// MI455X (gfx1250) — matrix operand order repaired and checked by reading, not compiled, not run on hardware
//
#include <hip/hip_runtime.h>
#include <hip/hip_bf16.h>

// Problem constants (match reference)
#define B_  8
#define NS_ 8192
#define NG_ 2048
#define D_  64

typedef __attribute__((ext_vector_type(16))) __bf16 v16bf;
typedef __attribute__((ext_vector_type(8)))  float  v8f;
typedef __attribute__((ext_vector_type(4)))  float  f4;

// ---------------------------------------------------------------------------
// WMMA wrapper: D(f32 16x16) = A(16x32 bf16) * B(32x16 bf16) + C
// ---------------------------------------------------------------------------
__device__ __forceinline__ v8f wmma_bf16(v16bf a, v16bf b, v8f c) {
  return __builtin_amdgcn_wmma_f32_16x16x32_bf16(
      /*neg_a=*/false, a, /*neg_b=*/false, b,
      /*c_mod=*/(short)0, c, /*reuse_a=*/false, /*reuse_b=*/false);
}

// ---------------------------------------------------------------------------
// CDNA5 async global->LDS copy (ASYNCcnt) helpers.
// VDST = 32-bit LDS byte address (low 32 bits of generic pointer = LDS offset,
// ISA section 10.2 aperture rules); VADDR = 64-bit global address; 16B/lane.
// ---------------------------------------------------------------------------
__device__ __forceinline__ void async_b128(unsigned lds, const void* g) {
  asm volatile("global_load_async_to_lds_b128 %0, %1, off"
               :: "v"(lds), "v"((unsigned long long)(uintptr_t)g)
               : "memory");
}
__device__ __forceinline__ void async_wait0() {
  asm volatile("s_wait_asynccnt 0x0" ::: "memory");
}
// Stage one 64x128 bf16 chunk (padded ld=136) -- each thread copies 64B.
__device__ __forceinline__ void stage_chunk_async(__bf16* BsBuf,
                                                  const __bf16* gsrc,
                                                  int cn, int cs) {
  const unsigned l0 = (unsigned)(uintptr_t)(BsBuf + cn * 136 + cs * 32);
#pragma unroll
  for (int q = 0; q < 4; ++q)
    async_b128(l0 + 16u * q, gsrc + 8 * q);
}

// ---------------------------------------------------------------------------
// A fragment (16x32, bf16, row-major source, row stride `ld` elements).
// lane m = lane&15; lane-half adds k+=8; elems 0..7 -> k..k+7,
// elems 8..15 -> k+16..k+23. Two contiguous 16B loads per lane.
// ---------------------------------------------------------------------------
__device__ __forceinline__ v16bf load_a_bf16_rm(const __bf16* base, int ld,
                                                int kbase, int lane) {
  const int m  = lane & 15;
  const int ko = (lane >> 4) << 3;
  const __bf16* p = base + (long)m * ld + kbase + ko;
  v16bf r;
  ((f4*)&r)[0] = *(const f4*)(p);
  ((f4*)&r)[1] = *(const f4*)(p + 16);
  return r;
}

// Same fragment, but source is f32 (rbf_w / activations): 4x b128 loads,
// convert to bf16 in registers (keeps the big GEMM memory-bound).
__device__ __forceinline__ v16bf load_a_f32_cvt(const float* base, long ld,
                                                long kbase, int lane) {
  const int m  = lane & 15;
  const int ko = (lane >> 4) << 3;
  const float* p = base + (long)m * ld + kbase + ko;
  f4 x0 = ((const f4*)p)[0];
  f4 x1 = ((const f4*)p)[1];
  f4 x2 = ((const f4*)(p + 16))[0];
  f4 x3 = ((const f4*)(p + 16))[1];
  v16bf r;
  r[0]  = (__bf16)x0.x; r[1]  = (__bf16)x0.y; r[2]  = (__bf16)x0.z; r[3]  = (__bf16)x0.w;
  r[4]  = (__bf16)x1.x; r[5]  = (__bf16)x1.y; r[6]  = (__bf16)x1.z; r[7]  = (__bf16)x1.w;
  r[8]  = (__bf16)x2.x; r[9]  = (__bf16)x2.y; r[10] = (__bf16)x2.z; r[11] = (__bf16)x2.w;
  r[12] = (__bf16)x3.x; r[13] = (__bf16)x3.y; r[14] = (__bf16)x3.z; r[15] = (__bf16)x3.w;
  return r;
}

// ---------------------------------------------------------------------------
// B fragment (32x16, bf16) from a TRANSPOSED source T[n][k] (row stride ldk).
// lane-half adds k+=8; elems 0..7 -> k..k+7, elems 8..15 -> k+16..k+23; n = n0 + lane&15.
// k is two runs of 8 per lane -> two 16B (b128) loads. Works for global or LDS.
// ---------------------------------------------------------------------------
__device__ __forceinline__ v16bf load_b_tr(const __bf16* T, long ldk, int n0,
                                           long kbase, int lane) {
  const int  n = n0 + (lane & 15);
  const long k = kbase + (long)((lane >> 4) << 3);
  const __bf16* p = T + (long)n * ldk + k;
  v16bf r;
  ((f4*)&r)[0] = *(const f4*)(p);
  ((f4*)&r)[1] = *(const f4*)(p + 16);
  return r;
}

// C layout (16x16 f32): lane n = lane&15, VGPR i -> row m = i + 8*(lane>>4).
__device__ __forceinline__ void store_c_bf16_rm(__bf16* C, const v8f* acc,
                                                int lane) {
  const int nlo = lane & 15, hi = lane >> 4;
#pragma unroll
  for (int j = 0; j < 4; ++j)
#pragma unroll
    for (int i = 0; i < 8; ++i)
      C[(long)(i + 8 * hi) * 64 + j * 16 + nlo] = (__bf16)acc[j][i];
}

// ---------------------------------------------------------------------------
// Kernel 1: pre-linear  y = x @ W + b
// Writes bf16 row-major [B,N,64] and bf16 transposed [B,64,N].
// grid = (N/128, B), block = 256 (8 waves, 16 rows each).
// ---------------------------------------------------------------------------
__global__ __launch_bounds__(256) void pre_linear_kernel(
    const float* __restrict__ x, const float* __restrict__ W,
    const float* __restrict__ bias, __bf16* __restrict__ y_rm,
    __bf16* __restrict__ y_T, int N) {
  __shared__ __bf16 WT[64 * 64];  // W transposed: [n][k], bf16
  const int b = blockIdx.y, t = threadIdx.x;
  for (int i = t; i < 64 * 64; i += 256) {
    int n = i >> 6, k = i & 63;
    WT[n * 64 + k] = (__bf16)W[k * 64 + n];
  }
  __syncthreads();

  const int wave = t >> 5, lane = t & 31;
  const int m0 = blockIdx.x * 128 + wave * 16;
  const float* xb = x + ((long)b * N + m0) * 64;

  v8f acc[4] = {};
#pragma unroll
  for (int kt = 0; kt < 2; ++kt) {
    v16bf a = load_a_f32_cvt(xb, 64, kt * 32, lane);
#pragma unroll
    for (int j = 0; j < 4; ++j)
      acc[j] = wmma_bf16(a, load_b_tr(WT, 64, j * 16, kt * 32, lane), acc[j]);
  }

  const int nlo = lane & 15, hi = lane >> 4;
#pragma unroll
  for (int j = 0; j < 4; ++j) {
    const int col = j * 16 + nlo;
    const float bv = bias[col];
    __attribute__((aligned(16))) __bf16 tmp[8];
#pragma unroll
    for (int i = 0; i < 8; ++i) {
      float v = acc[j][i] + bv;
      tmp[i] = (__bf16)v;
      y_rm[((long)b * N + m0 + i + 8 * hi) * 64 + col] = (__bf16)v;
    }
    // transposed layout: 8 consecutive rows -> one 16B store
    __bf16* pT = y_T + (((long)b * 64) + col) * N + m0 + 8 * hi;
    *(f4*)pT = *(const f4*)tmp;
  }
}

// ---------------------------------------------------------------------------
// Kernel 2: xs_out = rbf_w @ xg_in   (A row-major f32, cvt->bf16 on load)
// Workgroup tile: M=256 (8 waves x 32 rows), full N=64.
// B k-chunks (64x128 bf16, padded ld=136) streamed into DOUBLE-BUFFERED LDS
// with async global->LDS DMA (ASYNCcnt), overlapped with WMMA compute.
// grid = (M/256, B), block = 256.
// ---------------------------------------------------------------------------
__global__ __launch_bounds__(256) void msg_gemm_rm_kernel(
    const float* __restrict__ A, const __bf16* __restrict__ BT,
    __bf16* __restrict__ C_rm, int M, int K) {
  __shared__ __bf16 Bs[2][64 * 136];  // [buf][n][k within chunk], ld = 136
  const int b = blockIdx.y, t = threadIdx.x;
  const int wave = t >> 5, lane = t & 31;
  const int m0 = blockIdx.x * 256 + wave * 32;  // wave owns rows m0..m0+31
  const float*  Ab = A + ((long)b * M + m0) * K;
  const __bf16* Bb = BT + (long)b * 64 * K;

  const int cn = t >> 2;  // 0..63 : staging row n
  const int cs = t & 3;   // 0..3  : 32-element segment
  const __bf16* bsrc = Bb + (long)cn * K + cs * 32;

  // prologue: stage chunk 0
  stage_chunk_async(Bs[0], bsrc, cn, cs);
  async_wait0();
  __syncthreads();

  v8f acc[8] = {};
  int cur = 0;
  for (int kc = 0; kc < K; kc += 128) {
    if (kc + 128 < K) {
      // launch DMA for next chunk into the other buffer (overlaps compute)
      stage_chunk_async(Bs[cur ^ 1], bsrc + kc + 128, cn, cs);
      // prefetch next A chunk (streams 512 MiB of rbf_w)
      const float* pf = Ab + (long)(lane & 15) * K + kc + 128;
      __builtin_prefetch(pf, 0, 0);
      __builtin_prefetch(pf + (long)16 * K, 0, 0);
    }
    // ---- 4 k-steps of 32 over the staged chunk ----
#pragma unroll
    for (int ks = 0; ks < 128; ks += 32) {
      v16bf a0 = load_a_f32_cvt(Ab, K, kc + ks, lane);
      v16bf a1 = load_a_f32_cvt(Ab + (long)16 * K, K, kc + ks, lane);
#pragma unroll
      for (int j = 0; j < 4; ++j) {
        v16bf bf = load_b_tr(Bs[cur], 136, j * 16, ks, lane);
        acc[j]     = wmma_bf16(a0, bf, acc[j]);
        acc[4 + j] = wmma_bf16(a1, bf, acc[4 + j]);
      }
    }
    async_wait0();    // my DMA into Bs[cur^1] landed
    __syncthreads();  // everyone done reading Bs[cur] / writing Bs[cur^1]
    cur ^= 1;
  }
  store_c_bf16_rm(C_rm + ((long)b * M + m0) * 64, acc, lane);
  store_c_bf16_rm(C_rm + ((long)b * M + m0 + 16) * 64, acc + 4, lane);
}

// ---------------------------------------------------------------------------
// Kernel 3: xg_out = rbf_w^T @ xs_in. A' = rbf_w^T built by LDS transpose
// (coalesced f32 tile loads [32 s][128 g] -> bf16 LDS [128 g][32 s], ld=40).
// B (xs_in_T) streamed through async double-buffered LDS like kernel 2.
// grid = (G/128, B), block = 256.
// ---------------------------------------------------------------------------
__global__ __launch_bounds__(256) void msg_gemm_tr_kernel(
    const float* __restrict__ A, const __bf16* __restrict__ BT,
    __bf16* __restrict__ C_rm, int S, int G) {
  __shared__ __bf16 At[128 * 40];     // [m in tile][k in step], ld = 40
  __shared__ __bf16 Bs[2][64 * 136];  // B chunks, ld = 136
  const int b = blockIdx.y, t = threadIdx.x;
  const int wave = t >> 5, lane = t & 31;
  const int g0 = blockIdx.x * 128;
  const float*  Ab = A + (long)b * S * G;
  const __bf16* Bb = BT + (long)b * 64 * S;

  const int lrow = t >> 3;  // 0..31 : s within 32-chunk (A staging)
  const int lseg = t & 7;   // 0..7  : 16-col segment   (A staging)
  const int cn   = t >> 2;  // 0..63 : B staging row n
  const int cs   = t & 3;   // 0..3  : B staging segment
  const __bf16* bsrc = Bb + (long)cn * S + cs * 32;

  stage_chunk_async(Bs[0], bsrc, cn, cs);
  async_wait0();
  __syncthreads();

  v8f acc[4] = {};
  int cur = 0;
  for (long kc = 0; kc < S; kc += 128) {
    if (kc + 128 < S)
      stage_chunk_async(Bs[cur ^ 1], bsrc + kc + 128, cn, cs);
#pragma unroll
    for (int ks = 0; ks < 128; ks += 32) {
      const long k = kc + ks;
      const float* src = Ab + (k + lrow) * G + g0 + lseg * 16;
      f4 u0 = ((const f4*)src)[0];
      f4 u1 = ((const f4*)src)[1];
      f4 u2 = ((const f4*)src)[2];
      f4 u3 = ((const f4*)src)[3];
      if (k + 32 < S)
        __builtin_prefetch(src + 32 * G, 0, 0);
      __syncthreads();  // previous step's LDS reads complete
      const int mb = lseg * 16;
      At[(mb +  0) * 40 + lrow] = (__bf16)u0.x;
      At[(mb +  1) * 40 + lrow] = (__bf16)u0.y;
      At[(mb +  2) * 40 + lrow] = (__bf16)u0.z;
      At[(mb +  3) * 40 + lrow] = (__bf16)u0.w;
      At[(mb +  4) * 40 + lrow] = (__bf16)u1.x;
      At[(mb +  5) * 40 + lrow] = (__bf16)u1.y;
      At[(mb +  6) * 40 + lrow] = (__bf16)u1.z;
      At[(mb +  7) * 40 + lrow] = (__bf16)u1.w;
      At[(mb +  8) * 40 + lrow] = (__bf16)u2.x;
      At[(mb +  9) * 40 + lrow] = (__bf16)u2.y;
      At[(mb + 10) * 40 + lrow] = (__bf16)u2.z;
      At[(mb + 11) * 40 + lrow] = (__bf16)u2.w;
      At[(mb + 12) * 40 + lrow] = (__bf16)u3.x;
      At[(mb + 13) * 40 + lrow] = (__bf16)u3.y;
      At[(mb + 14) * 40 + lrow] = (__bf16)u3.z;
      At[(mb + 15) * 40 + lrow] = (__bf16)u3.w;
      __syncthreads();
      v16bf a = load_a_bf16_rm(At + (wave * 16) * 40, 40, 0, lane);
#pragma unroll
      for (int j = 0; j < 4; ++j)
        acc[j] = wmma_bf16(a, load_b_tr(Bs[cur], 136, j * 16, ks, lane), acc[j]);
    }
    async_wait0();
    __syncthreads();
    cur ^= 1;
  }
  store_c_bf16_rm(C_rm + ((long)b * G + g0) * 64, acc, lane);
}

// ---------------------------------------------------------------------------
// Kernel 4: post-linear  y = [x_in, x_out] @ Wp + bp  (K = 128), f32 output.
// Accumulators routed through a padded per-wave LDS tile so the final
// 20 MiB f32 output is written with coalesced b128 stores.
// grid = (M/128, B), block = 256.
// ---------------------------------------------------------------------------
__global__ __launch_bounds__(256) void post_linear_kernel(
    const __bf16* __restrict__ x_in, const __bf16* __restrict__ x_out,
    const float* __restrict__ Wp, const float* __restrict__ bias,
    float* __restrict__ out, int M, int rowOff) {
  __shared__ __bf16 WT[64 * 128];     // Wp transposed: [n][k0..127]
  __shared__ float  Cs[8 * 16 * 68];  // per-wave 16x64 tile, row pad 4 dwords
  const int b = blockIdx.y, t = threadIdx.x;
  for (int i = t; i < 64 * 128; i += 256) {
    int n = i >> 7, kk = i & 127;
    WT[n * 128 + kk] = (__bf16)Wp[kk * 64 + n];
  }
  __syncthreads();

  const int wave = t >> 5, lane = t & 31;
  const int m0 = blockIdx.x * 128 + wave * 16;
  const __bf16* A1 = x_in  + ((long)b * M + m0) * 64;
  const __bf16* A2 = x_out + ((long)b * M + m0) * 64;

  v8f acc[4] = {};
#pragma unroll
  for (int kt = 0; kt < 4; ++kt) {
    v16bf a = load_a_bf16_rm(kt < 2 ? A1 : A2, 64, (kt & 1) * 32, lane);
#pragma unroll
    for (int j = 0; j < 4; ++j)
      acc[j] = wmma_bf16(a, load_b_tr(WT, 128, j * 16, kt * 32, lane), acc[j]);
  }

  // bias + transpose-through-LDS (conflict-free write pattern), coalesced out
  const int nlo = lane & 15, hi = lane >> 4;
  float* cw = &Cs[wave * 16 * 68];
#pragma unroll
  for (int j = 0; j < 4; ++j) {
    const float bv = bias[j * 16 + nlo];
#pragma unroll
    for (int i = 0; i < 8; ++i)
      cw[(i + 8 * hi) * 68 + j * 16 + nlo] = acc[j][i] + bv;
  }
  __syncthreads();
  const int r = lane >> 1, sseg = (lane & 1) * 32;
  const float* rp = cw + r * 68 + sseg;
  float* ob = out + (((long)b * (NS_ + NG_)) + rowOff + m0 + r) * 64 + sseg;
#pragma unroll
  for (int q = 0; q < 8; ++q)
    ((f4*)ob)[q] = ((const f4*)rp)[q];
}

// ---------------------------------------------------------------------------
// Host-side launch
// ---------------------------------------------------------------------------
extern "C" void kernel_launch(void* const* d_in, const int* in_sizes, int n_in,
                              void* d_out, int out_size, void* d_ws,
                              size_t ws_size, hipStream_t stream) {
  const float* s_x   = (const float*)d_in[0];
  const float* g_x   = (const float*)d_in[1];
  const float* rbf_w = (const float*)d_in[2];
  const float* Ws    = (const float*)d_in[3];
  const float* bs    = (const float*)d_in[4];
  const float* Wg    = (const float*)d_in[5];
  const float* bg    = (const float*)d_in[6];
  const float* Wps   = (const float*)d_in[7];
  const float* bps   = (const float*)d_in[8];
  const float* Wpg   = (const float*)d_in[9];
  const float* bpg   = (const float*)d_in[10];
  float* out = (float*)d_out;

  // Workspace layout (bf16), ~30 MiB total
  char* ws = (char*)d_ws;
  __bf16* xs_in_rm  = (__bf16*)(ws);                // 8 MiB
  __bf16* xs_in_T   = (__bf16*)(ws + (8u  << 20));  // 8 MiB
  __bf16* xg_in_rm  = (__bf16*)(ws + (16u << 20));  // 2 MiB
  __bf16* xg_in_T   = (__bf16*)(ws + (18u << 20));  // 2 MiB
  __bf16* xs_out_rm = (__bf16*)(ws + (20u << 20));  // 8 MiB
  __bf16* xg_out_rm = (__bf16*)(ws + (28u << 20));  // 2 MiB

  dim3 blk(256);
  // 1) pre-linears
  pre_linear_kernel<<<dim3(NS_ / 128, B_), blk, 0, stream>>>(
      s_x, Ws, bs, xs_in_rm, xs_in_T, NS_);
  pre_linear_kernel<<<dim3(NG_ / 128, B_), blk, 0, stream>>>(
      g_x, Wg, bg, xg_in_rm, xg_in_T, NG_);
  // 2) big bipartite GEMMs (memory-bound on rbf_w)
  msg_gemm_rm_kernel<<<dim3(NS_ / 256, B_), blk, 0, stream>>>(
      rbf_w, xg_in_T, xs_out_rm, NS_, NG_);
  msg_gemm_tr_kernel<<<dim3(NG_ / 128, B_), blk, 0, stream>>>(
      rbf_w, xs_in_T, xg_out_rm, NS_, NG_);
  // 3) post-linears -> final f32 output [B, NS+NG, 64]
  post_linear_kernel<<<dim3(NS_ / 128, B_), blk, 0, stream>>>(
      xs_in_rm, xs_out_rm, Wps, bps, out, NS_, 0);
  post_linear_kernel<<<dim3(NG_ / 128, B_), blk, 0, stream>>>(
      xg_in_rm, xg_out_rm, Wpg, bpg, out, NG_, NS_);
}
